// BHEP_M_85701777424450
// MI455X (gfx1250) — matrix operand order repaired and checked by reading, not compiled, not run on hardware
//
#include <hip/hip_runtime.h>
#include <hip/hip_bf16.h>
#include <math.h>
#include <stdint.h>

// ---------------------------------------------------------------------------
// Gaussian pair-energy:
//   prep:  xbf = bf16(x) (one-shot, L2-resident 16MB), norms[i] = ||x_i||^2
//   tiles: S1 partials via bf16 WMMA Gram tiles + exp epilogue
//          (upper block triangle only; off-diag blocks weighted 2x)
//   finalize: deterministic fixed-order reduction + scales
// ---------------------------------------------------------------------------

typedef __attribute__((ext_vector_type(16))) __bf16 v16bf;
typedef __attribute__((ext_vector_type(8)))  __bf16 bf16x8;
typedef __attribute__((ext_vector_type(4)))  __bf16 bf16x4;
typedef __attribute__((ext_vector_type(8)))  float  v8f;

#define N_PTS   16384
#define DDIM    512
#define BETA_F  10.0f
#define TILE    128
#define KC      32
#define LDSW    40            // KC + 8 bf16 pad (16B, keeps b128 alignment)
#define NBLK    ((N_PTS/TILE)*(N_PTS/TILE))

#define USE_ASYNC_STAGE 1     // gfx1250 GLOBAL_LOAD_ASYNC_TO_LDS_B128 path

#if USE_ASYNC_STAGE
// GVS mode: mem = SADDR(64b sgpr) + VADDR(32b vgpr) + IOFFSET; the instruction
// writes LDS[vdst + IOFFSET], so offsets are folded into both operands here.
__device__ __forceinline__ void async_ld128(uint32_t lds_addr, uint64_t gbase,
                                            uint32_t goff) {
  asm volatile("global_load_async_to_lds_b128 %0, %1, %2 offset:0"
               :: "v"(lds_addr), "v"(goff), "s"(gbase) : "memory");
}
__device__ __forceinline__ void wait_async0() {
  asm volatile("s_wait_asynccnt 0x0" ::: "memory");
}
#endif

// ---------------------- prep: bf16 copy + norms -----------------------------
__global__ __launch_bounds__(256) void bhep_prep(const float* __restrict__ x,
                                                 __bf16* __restrict__ xbf,
                                                 float* __restrict__ norms) {
  const int wave = threadIdx.x >> 5, lane = threadIdx.x & 31;
  const int row  = blockIdx.x * 8 + wave;
  const float4* r = (const float4*)(x + (size_t)row * DDIM);
  __bf16* w = xbf + (size_t)row * DDIM;
  float s = 0.f;
  #pragma unroll
  for (int j = 0; j < 4; ++j) {
    float4 v = r[lane + 32 * j];
    s += v.x * v.x + v.y * v.y + v.z * v.z + v.w * v.w;
    bf16x4 q = { (__bf16)v.x, (__bf16)v.y, (__bf16)v.z, (__bf16)v.w };
    *(bf16x4*)(w + (size_t)(lane + 32 * j) * 4) = q;   // coalesced 8B stores
  }
  #pragma unroll
  for (int off = 16; off > 0; off >>= 1) s += __shfl_xor(s, off, 32);
  if (lane == 0) norms[row] = s;
}

// ---------------------- main tile kernel (WMMA) -----------------------------
__global__ __launch_bounds__(256) void bhep_tiles(const __bf16* __restrict__ xbf,
                                                  const float* __restrict__ norms,
                                                  float* __restrict__ partial) {
  const int ib = blockIdx.y, jb = blockIdx.x;
  const int slot = ib * gridDim.x + jb;
  const int tid = threadIdx.x;
  if (jb < ib) {                      // symmetric: only upper block triangle
    if (tid == 0) partial[slot] = 0.f;
    return;
  }

  __shared__ __bf16 As[2][TILE][LDSW];
  __shared__ __bf16 Bs[2][TILE][LDSW];
  __shared__ float  NsI[TILE], NsJ[TILE];
  __shared__ float  red[256];

  const int lane = tid & 31, wave = tid >> 5;
  const int waveRow = wave >> 2;      // 0..1  -> 64-row band
  const int waveCol = wave & 3;       // 0..3  -> 32-col band
  const int lr = lane & 15, lh = lane >> 4;
  const int i0 = ib * TILE, j0 = jb * TILE;

  // Staging map: thread -> (row, 16-elt half); 2x b128 per side per chunk.
  const int srow = tid >> 1;
  const int c16  = (tid & 1) << 4;
#if USE_ASYNC_STAGE
  const uint64_t baseA = (uint64_t)(uintptr_t)(xbf + (size_t)i0 * DDIM);
  const uint64_t baseB = (uint64_t)(uintptr_t)(xbf + (size_t)j0 * DDIM);
  const uint32_t rowoff = (uint32_t)((srow * DDIM + c16) * sizeof(__bf16));
#endif

  if (tid < TILE) NsI[tid] = norms[i0 + tid];
  else            NsJ[tid - TILE] = norms[j0 + (tid - TILE)];

  v8f acc[4][2];
  #pragma unroll
  for (int m = 0; m < 4; ++m)
    #pragma unroll
    for (int n = 0; n < 2; ++n)
      acc[m][n] = (v8f){0.f, 0.f, 0.f, 0.f, 0.f, 0.f, 0.f, 0.f};

#if USE_ASYNC_STAGE
  {
    // Preload chunk 0 into buffer 0.
    const uint32_t lA = (uint32_t)(uintptr_t)&As[0][srow][c16];
    const uint32_t lB = (uint32_t)(uintptr_t)&Bs[0][srow][c16];
    async_ld128(lA,      baseA, rowoff);
    async_ld128(lA + 16, baseA, rowoff + 16);
    async_ld128(lB,      baseB, rowoff);
    async_ld128(lB + 16, baseB, rowoff + 16);
  }
  wait_async0();
  __syncthreads();

  int buf = 0;
  for (int kc = 0; kc < DDIM; kc += KC) {
    // Prefetch next chunk into the other buffer while WMMAs run on this one.
    if (kc + KC < DDIM) {
      const int nb = buf ^ 1;
      const uint32_t go = rowoff + (uint32_t)((kc + KC) * sizeof(__bf16));
      const uint32_t lA = (uint32_t)(uintptr_t)&As[nb][srow][c16];
      const uint32_t lB = (uint32_t)(uintptr_t)&Bs[nb][srow][c16];
      async_ld128(lA,      baseA, go);
      async_ld128(lA + 16, baseA, go + 16);
      async_ld128(lB,      baseB, go);
      async_ld128(lB + 16, baseB, go + 16);
    }
#else
  int buf = 0;
  for (int kc = 0; kc < DDIM; kc += KC) {
    __syncthreads();
    { // synchronous bf16 tile copy (no conversion needed)
      uint4 va = *(const uint4*)(xbf + (size_t)(i0 + srow) * DDIM + kc + c16);
      uint4 vb = *(const uint4*)(xbf + (size_t)(j0 + srow) * DDIM + kc + c16);
      uint4 va2 = *(const uint4*)(xbf + (size_t)(i0 + srow) * DDIM + kc + c16 + 8);
      uint4 vb2 = *(const uint4*)(xbf + (size_t)(j0 + srow) * DDIM + kc + c16 + 8);
      *(uint4*)&As[buf][srow][c16]     = va;
      *(uint4*)&As[buf][srow][c16 + 8] = va2;
      *(uint4*)&Bs[buf][srow][c16]     = vb;
      *(uint4*)&Bs[buf][srow][c16 + 8] = vb2;
    }
    __syncthreads();
#endif

    // A fragments (16x32, M x K): lane lr = row, lh picks K octets.
    v16bf a[4];
    #pragma unroll
    for (int m = 0; m < 4; ++m) {
      const int row = waveRow * 64 + m * 16 + lr;
      bf16x8 lo = *(const bf16x8*)&As[buf][row][lh * 8];
      bf16x8 hi = *(const bf16x8*)&As[buf][row][lh * 8 + 16];
      a[m] = __builtin_shufflevector(lo, hi, 0,1,2,3,4,5,6,7,8,9,10,11,12,13,14,15);
    }
    // B fragments (32x16, K x N): lane lr = column j, lh picks K octets.
    v16bf b[2];
    #pragma unroll
    for (int n = 0; n < 2; ++n) {
      const int jr = waveCol * 32 + n * 16 + lr;
      bf16x8 lo = *(const bf16x8*)&Bs[buf][jr][lh * 8];
      bf16x8 hi = *(const bf16x8*)&Bs[buf][jr][lh * 8 + 16];
      b[n] = __builtin_shufflevector(lo, hi, 0,1,2,3,4,5,6,7,8,9,10,11,12,13,14,15);
    }

    #pragma unroll
    for (int m = 0; m < 4; ++m)
      #pragma unroll
      for (int n = 0; n < 2; ++n)
        acc[m][n] = __builtin_amdgcn_wmma_f32_16x16x32_bf16(
            false, a[m], false, b[n], (short)0, acc[m][n], false, false);

#if USE_ASYNC_STAGE
    wait_async0();        // prefetch landed (this wave)
    __syncthreads();      // ... and all waves' (frag ds_loads already waited
                          // before the WMMAs, so cur buffer is free to reuse)
    buf ^= 1;
#endif
  }

  // Epilogue: exp((2*dot + n_i + n_j)/(4B)); element e -> row e + 8*lh, col lr.
  const float inv4b = 1.0f / (4.0f * BETA_F);
  float psum = 0.f;
  #pragma unroll
  for (int n = 0; n < 2; ++n) {
    const float nj = NsJ[waveCol * 32 + n * 16 + lr];
    #pragma unroll
    for (int m = 0; m < 4; ++m) {
      const int ibase = waveRow * 64 + m * 16 + lh * 8;
      #pragma unroll
      for (int e = 0; e < 8; ++e) {
        const float ni = NsI[ibase + e];
        psum += __expf((2.f * acc[m][n][e] + ni + nj) * inv4b);
      }
    }
  }

  red[tid] = psum;
  __syncthreads();
  #pragma unroll
  for (int s = 128; s > 0; s >>= 1) {
    if (tid < s) red[tid] += red[tid + s];
    __syncthreads();
  }
  if (tid == 0) partial[slot] = (ib == jb ? 1.f : 2.f) * red[0];
}

// ---------------------- finalize (deterministic fixed-order reduce) ---------
__global__ __launch_bounds__(256) void bhep_finalize(const float* __restrict__ partial,
                                                     const float* __restrict__ norms,
                                                     float* __restrict__ out) {
  __shared__ float red[256];
  const int tid = threadIdx.x;

  float s1 = 0.f;
  for (int i = tid; i < NBLK; i += 256) s1 += partial[i];
  red[tid] = s1;
  __syncthreads();
  for (int s = 128; s > 0; s >>= 1) {
    if (tid < s) red[tid] += red[tid + s];
    __syncthreads();
  }
  const float S1 = red[0];
  __syncthreads();

  const float invr = 1.0f / (4.0f * BETA_F - 2.0f);
  float s2 = 0.f;
  for (int i = tid; i < N_PTS; i += 256) s2 += __expf(norms[i] * invr);
  red[tid] = s2;
  __syncthreads();
  for (int s = 128; s > 0; s >>= 1) {
    if (tid < s) red[tid] += red[tid + s];
    __syncthreads();
  }

  if (tid == 0) {
    const float S2 = red[0];
    // Double scales cast to float underflow to 0 exactly as the reference's
    // float32 arithmetic does (1e-256 / ~1e-250 -> 0.0f).
    const double lhs_scale = 1.0 / pow((double)BETA_F, (double)DDIM * 0.5);
    const double rhs_scale = 2.0 / pow((double)BETA_F - 0.5, (double)DDIM * 0.5);
    out[0] = (float)lhs_scale * (S1 / (float)N_PTS) - (float)rhs_scale * S2;
  }
}

// ---------------------------------------------------------------------------
extern "C" void kernel_launch(void* const* d_in, const int* in_sizes, int n_in,
                              void* d_out, int out_size, void* d_ws, size_t ws_size,
                              hipStream_t stream) {
  (void)in_sizes; (void)n_in; (void)out_size; (void)ws_size;
  const float* x = (const float*)d_in[0];
  __bf16* xbf    = (__bf16*)d_ws;                       // 16384*512 bf16 (16MB)
  float* norms   = (float*)(xbf + (size_t)N_PTS * DDIM);// 16384 floats
  float* partial = norms + N_PTS;                       // 16384 floats
  float* out     = (float*)d_out;

  bhep_prep<<<N_PTS / 8, 256, 0, stream>>>(x, xbf, norms);
  bhep_tiles<<<dim3(N_PTS / TILE, N_PTS / TILE), 256, 0, stream>>>(xbf, norms, partial);
  bhep_finalize<<<1, 256, 0, stream>>>(partial, norms, out);
}
